// RNNModel_88639535055426
// MI455X (gfx1250) — matrix operand order repaired and checked by reading, not compiled, not run on hardware
//
#include <hip/hip_runtime.h>
#include <math.h>

// ---------------------------------------------------------------------------
// Types for CDNA5 WMMA (wave32, 16x16x32 bf16 -> f32)
// ---------------------------------------------------------------------------
typedef __bf16 bf16_t;
typedef __attribute__((ext_vector_type(16))) __bf16 v16bf;
typedef __attribute__((ext_vector_type(8)))  __bf16 bf16x8;
typedef __attribute__((ext_vector_type(8)))  float   v8f;

// ---------------------------------------------------------------------------
// Problem constants (from reference)
// ---------------------------------------------------------------------------
constexpr int T_   = 48;
constexpr int B_   = 32;
constexpr int TB_  = T_ * B_;          // 1536
constexpr int NTOK = 32000;
constexpr int NINP = 1024;
constexpr int NHID = 2048;
constexpr int NB_  = 8;
constexpr int BS_  = 256;              // NHID / NB
constexpr int ATT_ = 1024;             // 4*BS
constexpr int DKIN = 64;
constexpr int GATES = 4 * BS_;         // 1024
constexpr int NGAT  = NB_ * GATES;     // 8192

__device__ __forceinline__ bf16_t f2bf(float f) {
  union { float f; unsigned u; } x; x.f = f;
  unsigned r = x.u + 0x7fffu + ((x.u >> 16) & 1u);   // round-nearest-even
  unsigned short h = (unsigned short)(r >> 16);
  return __builtin_bit_cast(bf16_t, h);
}
__device__ __forceinline__ float sigf(float x) { return 1.0f / (1.0f + __expf(-x)); }

// ---------------------------------------------------------------------------
// Generic batched bf16 GEMM:  C[z] = A[z] (MxK, row-major) * B[z]^T (NxK) + bias
//
// Fragment layouts (ISA 16-bit A 16x32 / B-equivalent, wave32):
//   A: lane<16 : M=lane,    K = {k..k+7, k+16..k+23}
//      lane>=16: M=lane-16, K = {k+8..k+15, k+24..k+31}
//   B (pre-transposed, N-major [N x K]):
//      lane<16 : N=lane,    K = {k..k+7, k+16..k+23}
//      lane>=16: N=lane-16, K = {k+8..k+15, k+24..k+31}
//
// K is a compile-time constant (KT) and the K-loop is fully unrolled: per-lane
// pointers are computed ONCE and every load uses a constant immediate offset
// (max 2*KT bytes, fits the 24-bit VMEM offset field) -> zero per-iteration
// address arithmetic, no evolving 64-bit bases to spill.
// Wave computes (MT*16) rows x 16 cols; MT accumulators share one B fragment;
// two named buffer sets alternate across 64-wide K-chunks (no rotation movs).
// Requires M % (MT*16)==0, N % 64==0, KT % 64==0.
// Block = 128 threads (4 waves, one 16-col slice each).
// ---------------------------------------------------------------------------
template <int MT, int KT>
__global__ void gemm_bf16_wmma(const bf16_t* __restrict__ A, int lda, long long sA,
                               const bf16_t* __restrict__ Bt, int ldb, long long sB,
                               float* Cf, bf16_t* Cb, int ldc, long long sC,
                               const float* bias, long long sBias,
                               int M, int N) {
  const int lane = threadIdx.x & 31;
  const int wave = threadIdx.x >> 5;
  const int n0 = blockIdx.x * 64 + wave * 16;
  const int m0 = blockIdx.y * (MT * 16);
  const int z  = blockIdx.z;
  if (n0 >= N || m0 >= M) return;

  A  += (size_t)z * sA;
  Bt += (size_t)z * sB;

  // per-lane pointers, computed once; K offsets are immediates after unroll
  const int la   = lane & 15;
  const int lh8  = (lane >= 16) ? 8 : 0;
  const int lh16 = (lane >= 16) ? 16 : 0;
  const bf16_t* aP[MT];
#pragma unroll
  for (int i = 0; i < MT; ++i)
    aP[i] = A + (size_t)(m0 + i * 16 + la) * lda + lh8;
  const bf16_t* bP = Bt + (size_t)(n0 + la) * ldb + lh8;

  v8f acc[MT];
#pragma unroll
  for (int i = 0; i < MT; ++i) acc[i] = (v8f){};

  v16bf a0[MT], a1[MT], b0, b1;

  auto loadTile = [&](v16bf (&af)[MT], v16bf& bfr, int k) {
    bf16x8 blo = *(const bf16x8*)(bP + k);
    bf16x8 bhi = *(const bf16x8*)(bP + k + 16);
#pragma unroll
    for (int e = 0; e < 8; ++e) { bfr[e] = blo[e]; bfr[e + 8] = bhi[e]; }
#pragma unroll
    for (int i = 0; i < MT; ++i) {
      bf16x8 lo = *(const bf16x8*)(aP[i] + k);
      bf16x8 hi = *(const bf16x8*)(aP[i] + k + 16);
#pragma unroll
      for (int e = 0; e < 8; ++e) { af[i][e] = lo[e]; af[i][e + 8] = hi[e]; }
    }
  };

  loadTile(a0, b0, 0);
#pragma unroll
  for (int k0 = 0; k0 < KT; k0 += 64) {
    // second half-tile of this chunk (always in range: KT % 64 == 0)
    loadTile(a1, b1, k0 + 32);
    if (k0 + 128 < KT) {   // global_prefetch_b8 two chunks ahead
      __builtin_prefetch(bP + k0 + 128, 0, 3);
      __builtin_prefetch(aP[0] + k0 + 128, 0, 3);
    }
#pragma unroll
    for (int i = 0; i < MT; ++i)
      acc[i] = __builtin_amdgcn_wmma_f32_16x16x32_bf16(false, a0[i], false, b0,
                                                       (short)0, acc[i], false, false);
    if (k0 + 64 < KT) loadTile(a0, b0, k0 + 64);
#pragma unroll
    for (int i = 0; i < MT; ++i)
      acc[i] = __builtin_amdgcn_wmma_f32_16x16x32_bf16(false, a1[i], false, b1,
                                                       (short)0, acc[i], false, false);
  }

  const int col = n0 + la;
  float bv = 0.0f;
  if (bias) bv = bias[(size_t)z * sBias + col];
  if (Cf) {
    float* C = Cf + (size_t)z * sC;
#pragma unroll
    for (int i = 0; i < MT; ++i) {
      const int rb = m0 + i * 16 + ((lane >= 16) ? 8 : 0);
#pragma unroll
      for (int r = 0; r < 8; ++r)
        C[(size_t)(rb + r) * ldc + col] = acc[i][r] + bv;
    }
  }
  if (Cb) {
    bf16_t* C = Cb + (size_t)z * sC;
#pragma unroll
    for (int i = 0; i < MT; ++i) {
      const int rb = m0 + i * 16 + ((lane >= 16) ? 8 : 0);
#pragma unroll
      for (int r = 0; r < 8; ++r)
        C[(size_t)(rb + r) * ldc + col] = f2bf(acc[i][r] + bv);
    }
  }
}

// ---------------------------------------------------------------------------
// Transpose f32 [R x C] -> bf16 [C x R] (N-major weight prep), batched in z.
// ---------------------------------------------------------------------------
__global__ void transpose_f32_bf16(const float* __restrict__ in, bf16_t* __restrict__ out,
                                   int R, int C, long long sIn, long long sOut) {
  __shared__ float tile[32][33];
  in  += (size_t)blockIdx.z * sIn;
  out += (size_t)blockIdx.z * sOut;
  const int c0 = blockIdx.x * 32, r0 = blockIdx.y * 32;
  const int tx = threadIdx.x & 31, ty = threadIdx.x >> 5;   // 256 thr: ty 0..7
#pragma unroll
  for (int i = 0; i < 32; i += 8) {
    int r = r0 + ty + i, c = c0 + tx;
    tile[ty + i][tx] = (r < R && c < C) ? in[(size_t)r * C + c] : 0.0f;
  }
  __syncthreads();
#pragma unroll
  for (int i = 0; i < 32; i += 8) {
    int r = r0 + tx, c = c0 + ty + i;
    if (c < C && r < R) out[(size_t)c * R + r] = f2bf(tile[tx][ty + i]);
  }
}

// ---------------------------------------------------------------------------
// Embedding gather -> bf16 rows [TB x NINP]
// ---------------------------------------------------------------------------
__global__ void gather_emb(const int* __restrict__ tok, const float* __restrict__ emb,
                           bf16_t* __restrict__ xb) {
  const int row = blockIdx.x;                 // t*B + b
  const int idx = tok[row];
  const float* src = emb + (size_t)idx * NINP;
  bf16_t* dst = xb + (size_t)row * NINP;
#pragma unroll
  for (int j = 0; j < 4; ++j) {
    int c = threadIdx.x * 4 + j;
    dst[c] = f2bf(src[c]);
  }
}

// biasih[n,g] = sum_d fc_in_b[d] * W_ih[n,d,g]
__global__ void bias_ih_kernel(const float* __restrict__ fcb, const float* __restrict__ Wih,
                               float* __restrict__ biasih) {
  const int n = blockIdx.y;
  const int g = blockIdx.x * 256 + threadIdx.x;
  float s = 0.0f;
  const float* w = Wih + ((size_t)n * ATT_) * GATES + g;
  for (int d = 0; d < ATT_; ++d) s += fcb[d] * w[(size_t)d * GATES];
  biasih[(size_t)n * GATES + g] = s;
}

// init recurrent state (f32 + bf16 shadow)
__global__ void init_state(const float* __restrict__ hx, const float* __restrict__ cx,
                           float* h0, float* c0, bf16_t* hbf) {
  const int i = blockIdx.x * 256 + threadIdx.x;
  float h = hx[i];
  h0[i] = h; c0[i] = cx[i]; hbf[i] = f2bf(h);
}

// ---------------------------------------------------------------------------
// Input attention score: s1[b,n] = (h_b[b,n,:] @ q_w[n]) . k1[t*B+b,:] / sqrt(64)
// Block = one (b,n), 256 threads (one per d).
// ---------------------------------------------------------------------------
__global__ void rim_score(int t, const float* __restrict__ h, const float* __restrict__ q_w,
                          const float* __restrict__ k1, float* __restrict__ s1) {
  const int bn = blockIdx.x;
  const int b = bn >> 3, n = bn & 7;
  const int tid = threadIdx.x;
  __shared__ float k1s[DKIN];
  __shared__ float red[256];
  if (tid < DKIN) k1s[tid] = k1[(size_t)(t * B_ + b) * DKIN + tid];
  __syncthreads();
  const float hv = h[(size_t)b * NHID + n * BS_ + tid];
  const float* qr = q_w + ((size_t)n * BS_ + tid) * DKIN;
  float acc = 0.0f;
#pragma unroll
  for (int k = 0; k < DKIN; ++k) acc += qr[k] * k1s[k];
  red[tid] = hv * acc;
  __syncthreads();
  for (int s = 128; s > 0; s >>= 1) {
    if (tid < s) red[tid] += red[tid + s];
    __syncthreads();
  }
  if (tid == 0) s1[bn] = red[0] * 0.125f;   // 1/sqrt(64)
}

// Gate a=sigmoid(s1) and top-k block mask (mask off the 3 smallest s1,
// stable tie-break by lowest index == jax.lax.top_k on null_att).
__global__ void rim_mask(const float* __restrict__ s1, float* __restrict__ a,
                         float* __restrict__ msk) {
  const int b = threadIdx.x;
  if (b >= B_) return;
  float v[NB_]; bool tk[NB_];
#pragma unroll
  for (int n = 0; n < NB_; ++n) {
    v[n] = s1[b * NB_ + n];
    a[b * NB_ + n] = sigf(v[n]);
    tk[n] = false;
  }
#pragma unroll
  for (int it = 0; it < NB_ - 5; ++it) {   // NB - TOPK = 3
    int best = -1;
#pragma unroll
    for (int n = 0; n < NB_; ++n)
      if (!tk[n] && (best < 0 || v[n] < v[best])) best = n;
    tk[best] = true;
  }
#pragma unroll
  for (int n = 0; n < NB_; ++n) msk[b * NB_ + n] = tk[n] ? 0.0f : 1.0f;
}

// ---------------------------------------------------------------------------
// Fused per-step update: LSTM pointwise + communication attention + merge.
// Block = one batch element b, 256 threads.
// gatesHH already holds h@W_hh + b_lstm; U holds vfc@W_ih (token-only, hoisted).
// ---------------------------------------------------------------------------
__global__ void rim_update(int t,
                           const float* __restrict__ hcur, const float* __restrict__ ccur,
                           const float* __restrict__ gatesHH,
                           const float* __restrict__ U,
                           const float* __restrict__ biasih,
                           const float* __restrict__ agate,
                           const float* __restrict__ maskm,
                           const float* __restrict__ cq_w, const float* __restrict__ ck_w,
                           const float* __restrict__ cv_w,
                           const float* __restrict__ cfc_w, const float* __restrict__ cfc_b,
                           float* __restrict__ hout, float* __restrict__ cout_,
                           bf16_t* __restrict__ hbf, bf16_t* __restrict__ outs_bf) {
  const int b = blockIdx.x;
  const int tid = threadIdx.x;
  __shared__ float hn[NHID];          // h_new per block (8 RIM blocks x 256)
  __shared__ float cn[NHID];
  __shared__ float qc[512], kc[512], vc[512];   // [n][h*16+k]
  __shared__ float csr[256], cat[256];          // [h*64 + n*8 + m]
  __shared__ float co[512];                     // [n][h*16+v]

  const float* Urow = U + (size_t)(t * B_ + b) * NGAT;
  const float* Grow = gatesHH + (size_t)b * NGAT;

  // 1) LSTM gates + cell update
#pragma unroll
  for (int i = 0; i < 8; ++i) {
    const int o = tid + i * 256;            // 0..2047
    const int n = o >> 8, s = o & 255;
    const float av = agate[b * NB_ + n];
    const size_t gb = (size_t)n * GATES;
    const float ig = av * Urow[gb + s]       + biasih[gb + s]       + Grow[gb + s];
    const float fg = av * Urow[gb + 256 + s] + biasih[gb + 256 + s] + Grow[gb + 256 + s];
    const float gg = av * Urow[gb + 512 + s] + biasih[gb + 512 + s] + Grow[gb + 512 + s];
    const float og = av * Urow[gb + 768 + s] + biasih[gb + 768 + s] + Grow[gb + 768 + s];
    const float cb = ccur[(size_t)b * NHID + o];
    const float cnv = sigf(fg) * cb + sigf(ig) * tanhf(gg);
    cn[o] = cnv;
    hn[o] = sigf(og) * tanhf(cnv);
  }
  __syncthreads();

  // 2) qc/kc/vc projections (per-block 256->64)
#pragma unroll
  for (int i = 0; i < 2; ++i) {
    const int o = tid + i * 256;            // 0..511: n = o>>6, j = o&63
    const int n = o >> 6, j = o & 63;
    const float* hb = &hn[n * BS_];
    float aq = 0.0f, ak = 0.0f, av = 0.0f;
    const size_t wb = (size_t)n * BS_ * 64 + j;
    for (int d = 0; d < BS_; ++d) {
      const float hv = hb[d];
      aq += hv * cq_w[wb + (size_t)d * 64];
      ak += hv * ck_w[wb + (size_t)d * 64];
      av += hv * cv_w[wb + (size_t)d * 64];
    }
    qc[o] = aq; kc[o] = ak; vc[o] = av;
  }
  __syncthreads();

  // 3) attention scores over blocks: cs[h,n,m], softmax over m
  {
    const int h = tid >> 6, n = (tid >> 3) & 7, m = tid & 7;
    float s = 0.0f;
#pragma unroll
    for (int k = 0; k < 16; ++k) s += qc[n * 64 + h * 16 + k] * kc[m * 64 + h * 16 + k];
    csr[tid] = s * 0.25f;                   // 1/sqrt(16)
  }
  __syncthreads();
  {
    const int base = tid & ~7;
    float mx = csr[base];
#pragma unroll
    for (int m = 1; m < 8; ++m) mx = fmaxf(mx, csr[base + m]);
    float sm = 0.0f;
#pragma unroll
    for (int m = 0; m < 8; ++m) sm += __expf(csr[base + m] - mx);
    cat[tid] = __expf(csr[tid] - mx) / sm;
  }
  __syncthreads();

  // 4) co[n,h,v] = sum_m catt[h,n,m] * vc[m,h,v]
#pragma unroll
  for (int i = 0; i < 2; ++i) {
    const int o = tid + i * 256;
    const int n = o >> 6, hv = o & 63, h = hv >> 4, v = hv & 15;
    float s = 0.0f;
#pragma unroll
    for (int m = 0; m < 8; ++m) s += cat[h * 64 + n * 8 + m] * vc[m * 64 + h * 16 + v];
    co[o] = s;
  }
  __syncthreads();

  // 5) h_new += co @ cfc_w + cfc_b; masked merge; emit f32 state + bf16 shadows
#pragma unroll
  for (int i = 0; i < 8; ++i) {
    const int o = tid + i * 256;
    const int n = o >> 8, s = o & 255;
    float add = cfc_b[s];
    const float* con = &co[n * 64];
    for (int k = 0; k < 64; ++k) add += con[k] * cfc_w[(size_t)k * BS_ + s];
    const float h2 = hn[o] + add;
    const float mk = maskm[b * NB_ + n];
    const float hb0 = hcur[(size_t)b * NHID + o];
    const float cb0 = ccur[(size_t)b * NHID + o];
    const float ho = mk * h2    + (1.0f - mk) * hb0;
    const float cv = mk * cn[o] + (1.0f - mk) * cb0;
    hout[(size_t)b * NHID + o] = ho;
    cout_[(size_t)b * NHID + o] = cv;
    hbf[(size_t)b * NHID + o] = f2bf(ho);
    outs_bf[(size_t)(t * B_ + b) * NHID + o] = f2bf(ho);
  }
}

// copy final hx/cx into the output tail
__global__ void final_state(const float* __restrict__ h, const float* __restrict__ c,
                            float* __restrict__ out) {
  const int i = blockIdx.x * 256 + threadIdx.x;
  const size_t base = (size_t)T_ * B_ * NTOK;
  out[base + i] = h[i];
  out[base + (size_t)B_ * NHID + i] = c[i];
}

// ===========================================================================
extern "C" void kernel_launch(void* const* d_in, const int* in_sizes, int n_in,
                              void* d_out, int out_size, void* d_ws, size_t ws_size,
                              hipStream_t stream) {
  (void)in_sizes; (void)n_in; (void)out_size; (void)ws_size;

  const int*   tok     = (const int*)d_in[0];
  const float* hx      = (const float*)d_in[1];
  const float* cx      = (const float*)d_in[2];
  const float* emb     = (const float*)d_in[3];
  const float* q_w     = (const float*)d_in[4];
  const float* k_w     = (const float*)d_in[5];
  const float* v_w     = (const float*)d_in[6];
  const float* fc_in_w = (const float*)d_in[7];
  const float* fc_in_b = (const float*)d_in[8];
  const float* W_ih    = (const float*)d_in[9];
  const float* W_hh    = (const float*)d_in[10];
  const float* b_lstm  = (const float*)d_in[11];
  const float* cq_w    = (const float*)d_in[12];
  const float* ck_w    = (const float*)d_in[13];
  const float* cv_w    = (const float*)d_in[14];
  const float* cfc_w   = (const float*)d_in[15];
  const float* cfc_b   = (const float*)d_in[16];
  const float* dec_w   = (const float*)d_in[17];
  const float* dec_b   = (const float*)d_in[18];
  float* logits = (float*)d_out;

  // ---- workspace bump allocator -------------------------------------------
  char* wp = (char*)d_ws;
  auto alloc = [&](size_t bytes) -> char* {
    char* r = wp;
    wp += (bytes + 255) & ~(size_t)255;
    return r;
  };
  bf16_t* xb     = (bf16_t*)alloc((size_t)TB_ * NINP * 2);
  bf16_t* kw_t   = (bf16_t*)alloc((size_t)DKIN * NINP * 2);
  bf16_t* vw_t   = (bf16_t*)alloc((size_t)ATT_ * NINP * 2);
  bf16_t* fcw_t  = (bf16_t*)alloc((size_t)ATT_ * ATT_ * 2);
  bf16_t* wih_t  = (bf16_t*)alloc((size_t)NB_ * GATES * ATT_ * 2);
  bf16_t* whh_t  = (bf16_t*)alloc((size_t)NB_ * GATES * BS_ * 2);
  bf16_t* decw_t = (bf16_t*)alloc((size_t)NTOK * NHID * 2);
  float*  k1     = (float*) alloc((size_t)TB_ * DKIN * 4);
  bf16_t* v1_b   = (bf16_t*)alloc((size_t)TB_ * ATT_ * 2);
  bf16_t* vfc_b  = (bf16_t*)alloc((size_t)TB_ * ATT_ * 2);
  float*  U      = (float*) alloc((size_t)TB_ * NGAT * 4);
  float*  biasih = (float*) alloc((size_t)NB_ * GATES * 4);
  float*  s1     = (float*) alloc((size_t)B_ * NB_ * 4);
  float*  agate  = (float*) alloc((size_t)B_ * NB_ * 4);
  float*  maskm  = (float*) alloc((size_t)B_ * NB_ * 4);
  float*  gates  = (float*) alloc((size_t)B_ * NGAT * 4);
  float*  h0     = (float*) alloc((size_t)B_ * NHID * 4);
  float*  c0     = (float*) alloc((size_t)B_ * NHID * 4);
  float*  h1     = (float*) alloc((size_t)B_ * NHID * 4);
  float*  c1     = (float*) alloc((size_t)B_ * NHID * 4);
  bf16_t* hbf    = (bf16_t*)alloc((size_t)B_ * NHID * 2);
  bf16_t* outsb  = (bf16_t*)alloc((size_t)TB_ * NHID * 2);

  // ---- phase 0: weight prep (f32 -> N-major bf16) --------------------------
  transpose_f32_bf16<<<dim3(2, 32, 1),    256, 0, stream>>>(k_w,     kw_t,  NINP, DKIN, 0, 0);
  transpose_f32_bf16<<<dim3(32, 32, 1),   256, 0, stream>>>(v_w,     vw_t,  NINP, ATT_, 0, 0);
  transpose_f32_bf16<<<dim3(32, 32, 1),   256, 0, stream>>>(fc_in_w, fcw_t, ATT_, ATT_, 0, 0);
  transpose_f32_bf16<<<dim3(32, 32, 8),   256, 0, stream>>>(W_ih,    wih_t, ATT_, GATES,
                                                            (long long)ATT_ * GATES,
                                                            (long long)GATES * ATT_);
  transpose_f32_bf16<<<dim3(32, 8, 8),    256, 0, stream>>>(W_hh,    whh_t, BS_,  GATES,
                                                            (long long)BS_ * GATES,
                                                            (long long)GATES * BS_);
  transpose_f32_bf16<<<dim3(1000, 64, 1), 256, 0, stream>>>(dec_w,   decw_t, NHID, NTOK, 0, 0);

  gather_emb<<<dim3(TB_), 256, 0, stream>>>(tok, emb, xb);
  bias_ih_kernel<<<dim3(4, 8), 256, 0, stream>>>(fc_in_b, W_ih, biasih);
  init_state<<<dim3(B_ * NHID / 256), 256, 0, stream>>>(hx, cx, h0, c0, hbf);

  // ---- phase 1: token-only GEMMs (hoisted out of the recurrence) -----------
  // k1 = x @ k_w                         (1536 x 64, K=1024)
  gemm_bf16_wmma<3, NINP><<<dim3(1, TB_ / 48, 1), 128, 0, stream>>>(
      xb, NINP, 0, kw_t, NINP, 0, k1, nullptr, DKIN, 0, nullptr, 0, TB_, DKIN);
  // v1 = x @ v_w                         (1536 x 1024, K=1024) -> bf16
  gemm_bf16_wmma<3, NINP><<<dim3(ATT_ / 64, TB_ / 48, 1), 128, 0, stream>>>(
      xb, NINP, 0, vw_t, NINP, 0, nullptr, v1_b, ATT_, 0, nullptr, 0, TB_, ATT_);
  // vfc = v1 @ fc_in_w                   (1536 x 1024, K=1024) -> bf16
  gemm_bf16_wmma<3, ATT_><<<dim3(ATT_ / 64, TB_ / 48, 1), 128, 0, stream>>>(
      v1_b, ATT_, 0, fcw_t, ATT_, 0, nullptr, vfc_b, ATT_, 0, nullptr, 0, TB_, ATT_);
  // U[n] = vfc @ W_ih[n]   batched over n (1536 x 1024 x 1024) x 8
  gemm_bf16_wmma<3, ATT_><<<dim3(GATES / 64, TB_ / 48, NB_), 128, 0, stream>>>(
      vfc_b, ATT_, 0, wih_t, ATT_, (long long)GATES * ATT_,
      U, nullptr, NGAT, (long long)GATES, nullptr, 0, TB_, GATES);

  // ---- phase 2: the serial recurrence (only truly recurrent work) ----------
  for (int t = 0; t < T_; ++t) {
    const float* hc = (t & 1) ? h1 : h0;
    const float* cc = (t & 1) ? c1 : c0;
    float* hn = (t & 1) ? h0 : h1;
    float* cn = (t & 1) ? c0 : c1;

    rim_score<<<dim3(B_ * NB_), 256, 0, stream>>>(t, hc, q_w, k1, s1);
    rim_mask<<<dim3(1), 32, 0, stream>>>(s1, agate, maskm);
    // gatesHH[n] = h_b[:,n,:] @ W_hh[n] + b_lstm[n]   (32 x 1024, K=256) x 8
    gemm_bf16_wmma<2, BS_><<<dim3(GATES / 64, 1, NB_), 128, 0, stream>>>(
        hbf, NHID, (long long)BS_, whh_t, BS_, (long long)GATES * BS_,
        gates, nullptr, NGAT, (long long)GATES, b_lstm, (long long)GATES,
        B_, GATES);
    rim_update<<<dim3(B_), 256, 0, stream>>>(
        t, hc, cc, gates, U, biasih, agate, maskm,
        cq_w, ck_w, cv_w, cfc_w, cfc_b, hn, cn, hbf, outsb);
  }

  // ---- phase 3: decoder GEMM (dominant 201 GFLOP) --------------------------
  // logits = outs @ dec_w + dec_b        (1536 x 32000, K=2048)
  gemm_bf16_wmma<3, NHID><<<dim3(NTOK / 64, TB_ / 48, 1), 128, 0, stream>>>(
      outsb, NHID, 0, decw_t, NHID, 0, logits, nullptr, NTOK, 0,
      dec_b, 0, TB_, NTOK);

  // final hx/cx (after 48 steps the state lands back in h0/c0)
  final_state<<<dim3(B_ * NHID / 256), 256, 0, stream>>>(h0, c0, logits);
}
